// LlamaDecoderLayerWithGNN_81269371174928
// MI455X (gfx1250) — compile-verified, matrix operand order checked by reading, not run on hardware
//
#include <hip/hip_runtime.h>

// ---------------------------------------------------------------------------
// Problem constants (match reference)
// ---------------------------------------------------------------------------
static constexpr int Bc = 2;
static constexpr int Sc = 1024;
static constexpr int Dc = 2048;
static constexpr int Hc = 32;
static constexpr int HDc = 64;
static constexpr int Ic = 8192;
static constexpr int AHc = 64;

typedef __attribute__((ext_vector_type(16))) __bf16 v16bf;
typedef __attribute__((ext_vector_type(8)))  float  v8f;
typedef __attribute__((ext_vector_type(4)))  int    v4i;

#define GLOBAL_AS __attribute__((address_space(1)))
#define LDS_AS    __attribute__((address_space(3)))

// ---- async global->LDS staging (CDNA5), guarded so absence still compiles ----
#if __has_builtin(__builtin_amdgcn_global_load_async_to_lds_b128)
#define ASYNC_LDS 1
__device__ __forceinline__ void async_b128(const unsigned short* g, unsigned short* l) {
    __builtin_amdgcn_global_load_async_to_lds_b128(
        (GLOBAL_AS v4i*)(unsigned short*)g, (LDS_AS v4i*)l, 0, 0);
}
#if __has_builtin(__builtin_amdgcn_s_wait_asynccnt)
#define WAIT_ASYNC(n) __builtin_amdgcn_s_wait_asynccnt(n)
#else
#define WAIT_ASYNC(n) asm volatile("s_wait_asynccnt %0" ::"n"(n))
#endif
#else
#define WAIT_ASYNC(n) do {} while (0)
#endif

// ---------------------------------------------------------------------------
// bf16 helpers (bit-level, round-to-nearest-even)
// ---------------------------------------------------------------------------
__device__ __forceinline__ unsigned short f2bf(float f) {
    unsigned u = __float_as_uint(f);
    u += 0x7FFFu + ((u >> 16) & 1u);
    return (unsigned short)(u >> 16);
}
__device__ __forceinline__ float bf2f(unsigned short h) {
    return __uint_as_float(((unsigned)h) << 16);
}
__device__ __forceinline__ float sigmoidf(float x) {
    return 1.0f / (1.0f + __expf(-x));
}

// ---------------------------------------------------------------------------
// Tile staging: nrows x 32 k of bf16 into LDS [row][k]; 256 threads stage
// 128 rows per pass, each thread moving 16 contiguous bf16 (32B).
// Returns #async instructions issued per wave (for deferred s_wait_asynccnt).
// ---------------------------------------------------------------------------
__device__ __forceinline__ int stage_tile(const unsigned short* __restrict__ Mtx,
                                          unsigned short* lds,
                                          int r0, int k0, int R, int ld,
                                          int tid, int nrows)
{
    int cnt = 0;
    #pragma unroll
    for (int sub = 0; sub < 256; sub += 128) {
        if (sub >= nrows) break;
        const int row  = sub + (tid >> 1);
        const int half = tid & 1;
        unsigned short* l = lds + row * 32 + half * 16;
#ifdef ASYNC_LDS
        if (r0 + sub + 128 <= R) {
            async_b128(Mtx + (long long)(r0 + row) * ld + k0 + half * 16, l);
            ++cnt;
            continue;
        }
#endif
        uint4 u0 = make_uint4(0u, 0u, 0u, 0u);
        uint4 u1 = make_uint4(0u, 0u, 0u, 0u);
        if (r0 + row < R) {
            const uint4* s4 = (const uint4*)(Mtx + (long long)(r0 + row) * ld + k0 + half * 16);
            u0 = s4[0];
            u1 = s4[1];
        }
        uint4* d4 = (uint4*)l;
        d4[0] = u0;
        d4[1] = u1;
    }
    return cnt;
}

// ---------------------------------------------------------------------------
// Batched bf16 GEMM, fp32 accumulate via v_wmma_f32_16x16x32_bf16
//   C[M,N] = alpha * A[M,K] @ B^T      (B stored [N,K] row-major)
// Block: 256 threads = 8 waves, tile 128(M) x 256(N), K-step 32,
// double-buffered LDS with async global->LDS staging.
// Wave grid 2(M) x 4(N); each wave computes 64x64 => 4x4 WMMA fragments
// (16 v_wmma per K-step against 16 ds_load_b128 -> 1.0 WMMA/load).
// cmode: 0 = dense; 1 = causal-skip tiles with n0 >= m0+128 (scores);
//        2 = limit K to m0+128 (attn @ V with causal-zero probs).
// ---------------------------------------------------------------------------
__global__ __launch_bounds__(256)
void k_gemm_bf16(const unsigned short* __restrict__ Ag,
                 const unsigned short* __restrict__ Bg,
                 float* __restrict__ Cg,
                 int M, int N, int K,
                 int lda, int ldb, int ldc,
                 long long strideA, long long strideB, long long strideC,
                 int cmode, float alpha)
{
    const int m0 = blockIdx.y * 128;
    const int n0 = blockIdx.x * 256;
    if (cmode == 1 && n0 >= m0 + 128) return;           // fully-masked tile
    const int Keff = (cmode == 2) ? (m0 + 128 < K ? m0 + 128 : K) : K;

    const int z = blockIdx.z;
    const unsigned short* A = Ag + (long long)z * strideA;
    const unsigned short* B = Bg + (long long)z * strideB;
    float* C = Cg + (long long)z * strideC;

    const int tid    = threadIdx.x;
    const int lane   = tid & 31;
    const int waveId = tid >> 5;
    const int wm     = waveId >> 2;    // 0..1 along M (64 rows each)
    const int wn     = waveId & 3;     // 0..3 along N (64 cols each)
    const int lane16 = lane & 15;
    const int lhalf  = lane >> 4;

    __shared__ __align__(16) unsigned short As[2][128 * 32];   // 2 x 8KB
    __shared__ __align__(16) unsigned short Bs[2][256 * 32];   // 2 x 16KB

    v8f acc[4][4] = {};

    int cur = 0;
    int pendNext = 0;
    stage_tile(A, As[0], m0, 0, M, lda, tid, 128);
    stage_tile(B, Bs[0], n0, 0, N, ldb, tid, 256);

    for (int k0 = 0; k0 < Keff; k0 += 32) {
        const int nxt = cur ^ 1;
        pendNext = 0;
        if (k0 + 32 < Keff) {
            pendNext  = stage_tile(A, As[nxt], m0, k0 + 32, M, lda, tid, 128);
            pendNext += stage_tile(B, Bs[nxt], n0, k0 + 32, N, ldb, tid, 256);
        }
        // wait for current tile's async copies; leave next tile's in flight
        if (pendNext >= 3)      { WAIT_ASYNC(3); }
        else if (pendNext == 2) { WAIT_ASYNC(2); }
        else if (pendNext == 1) { WAIT_ASYNC(1); }
        else                    { WAIT_ASYNC(0); }
        __syncthreads();

        union FragU { uint4 q[2]; v16bf v; };
        FragU af[4], bf[4];
        const uint4* a4 = (const uint4*)As[cur];
        const uint4* b4 = (const uint4*)Bs[cur];
        #pragma unroll
        for (int i = 0; i < 4; ++i) {
            const int row = wm * 64 + i * 16 + lane16;
            af[i].q[0] = a4[(row * 32 + lhalf * 8) >> 3];
            af[i].q[1] = a4[(row * 32 + 16 + lhalf * 8) >> 3];
        }
        #pragma unroll
        for (int j = 0; j < 4; ++j) {
            const int nrow = wn * 64 + j * 16 + lane16;
            bf[j].q[0] = b4[(nrow * 32 + lhalf * 8) >> 3];
            bf[j].q[1] = b4[(nrow * 32 + 16 + lhalf * 8) >> 3];
        }

        #pragma unroll
        for (int i = 0; i < 4; ++i)
            #pragma unroll
            for (int j = 0; j < 4; ++j)
                acc[i][j] = __builtin_amdgcn_wmma_f32_16x16x32_bf16(
                    false, af[i].v, false, bf[j].v, (short)0, acc[i][j], false, false);

        __syncthreads();
        cur = nxt;
    }

    #pragma unroll
    for (int i = 0; i < 4; ++i) {
        #pragma unroll
        for (int j = 0; j < 4; ++j) {
            #pragma unroll
            for (int r = 0; r < 8; ++r) {
                const int row = m0 + wm * 64 + i * 16 + lhalf * 8 + r;
                const int col = n0 + wn * 64 + j * 16 + lane16;
                if (row < M && col < N)
                    C[(long long)row * ldc + col] = acc[i][j][r] * alpha;
            }
        }
    }
}

// ---------------------------------------------------------------------------
// RMSNorm -> bf16 : one block (256 thr) per row of D elements
// ---------------------------------------------------------------------------
__global__ __launch_bounds__(256)
void k_rmsnorm_bf16(const float* __restrict__ x, const float* __restrict__ w,
                    unsigned short* __restrict__ out, int Dd)
{
    const int row = blockIdx.x;
    const int tid = threadIdx.x;
    const float* xr = x + (long long)row * Dd;
    __shared__ float red[256];
    float ss = 0.f;
    for (int i = tid; i < Dd; i += 256) { float v = xr[i]; ss += v * v; }
    red[tid] = ss; __syncthreads();
    for (int s = 128; s > 0; s >>= 1) {
        if (tid < s) red[tid] += red[tid + s];
        __syncthreads();
    }
    const float rs = rsqrtf(red[0] / (float)Dd + 1e-6f);
    unsigned short* orow = out + (long long)row * Dd;
    for (int i = tid; i < Dd; i += 256)
        orow[i] = f2bf(xr[i] * rs * w[i]);
}

// ---------------------------------------------------------------------------
// f32 -> bf16 cast (plain) and cast+transpose ([K,N] f32 -> [N,K] bf16)
// ---------------------------------------------------------------------------
__global__ void k_cast_bf16(const float* __restrict__ src,
                            unsigned short* __restrict__ dst, long long n)
{
    long long i = (long long)blockIdx.x * blockDim.x + threadIdx.x;
    if (i < n) dst[i] = f2bf(src[i]);
}
__global__ void k_cast_t_bf16(const float* __restrict__ src,
                              unsigned short* __restrict__ dst, int K, int N)
{
    long long i = (long long)blockIdx.x * blockDim.x + threadIdx.x;
    if (i >= (long long)K * N) return;
    const int k = (int)(i % K);
    const int n = (int)(i / K);
    dst[i] = f2bf(src[(long long)k * N + n]);   // dst[n*K + k]
}

// ---------------------------------------------------------------------------
// RoPE: thread per (b,h,s). q/k f32 [B,S,H*HD] -> bf16 [B,H,S,HD];
// v -> bf16 transposed [B,H,HD,S] so attn@V GEMM sees B as [N,K].
// ---------------------------------------------------------------------------
__global__ void k_rope(const float* __restrict__ qf, const float* __restrict__ kf,
                       const float* __restrict__ vf,
                       unsigned short* __restrict__ qb, unsigned short* __restrict__ kb,
                       unsigned short* __restrict__ vbt)
{
    const int idx = blockIdx.x * blockDim.x + threadIdx.x;
    const int total = Bc * Hc * Sc;
    if (idx >= total) return;
    const int s = idx % Sc;
    const int h = (idx / Sc) % Hc;
    const int b = idx / (Sc * Hc);
    const long long srcRow = ((long long)(b * Sc + s) * Hc + h) * HDc;
    const long long dstRow = (long long)idx * HDc;
    const long long vBase  = ((long long)(b * Hc + h) * HDc) * Sc + s;  // [B,H,HD,S]
    #pragma unroll 4
    for (int d = 0; d < HDc / 2; ++d) {
        const float inv = __powf(10000.0f, -(float)d / 32.0f);
        const float ang = (float)s * inv;
        const float c = __cosf(ang), sn = __sinf(ang);
        const float q1 = qf[srcRow + d],        q2 = qf[srcRow + d + 32];
        const float k1 = kf[srcRow + d],        k2 = kf[srcRow + d + 32];
        qb[dstRow + d]      = f2bf(q1 * c - q2 * sn);
        qb[dstRow + d + 32] = f2bf(q2 * c + q1 * sn);
        kb[dstRow + d]      = f2bf(k1 * c - k2 * sn);
        kb[dstRow + d + 32] = f2bf(k2 * c + k1 * sn);
        vbt[vBase + (long long)d * Sc]        = f2bf(vf[srcRow + d]);
        vbt[vBase + (long long)(d + 32) * Sc] = f2bf(vf[srcRow + d + 32]);
    }
}

// ---------------------------------------------------------------------------
// Causal softmax per (b,h,q) row. Writes probs [B,H,S,S] and probs_t [B,S,S,H]
// ---------------------------------------------------------------------------
__global__ __launch_bounds__(256)
void k_softmax_causal(const float* __restrict__ scores,
                      unsigned short* __restrict__ probs,
                      unsigned short* __restrict__ probs_t)
{
    const int row = blockIdx.x;           // b*H*S + h*S + q
    const int tid = threadIdx.x;
    const int q = row % Sc;
    const int h = (row / Sc) % Hc;
    const int b = row / (Sc * Hc);
    const int valid = q + 1;
    const float* srow = scores + (long long)row * Sc;
    __shared__ float red[256];

    float mx = -3.0e38f;
    for (int j = tid; j < valid; j += 256) mx = fmaxf(mx, srow[j]);
    red[tid] = mx; __syncthreads();
    for (int s = 128; s > 0; s >>= 1) {
        if (tid < s) red[tid] = fmaxf(red[tid], red[tid + s]);
        __syncthreads();
    }
    mx = red[0]; __syncthreads();

    float sum = 0.f;
    for (int j = tid; j < valid; j += 256) sum += __expf(srow[j] - mx);
    red[tid] = sum; __syncthreads();
    for (int s = 128; s > 0; s >>= 1) {
        if (tid < s) red[tid] += red[tid + s];
        __syncthreads();
    }
    const float inv = 1.0f / red[0];

    unsigned short* prow = probs + (long long)row * Sc;
    for (int j = tid; j < Sc; j += 256) {
        const float p = (j < valid) ? __expf(srow[j] - mx) * inv : 0.0f;
        const unsigned short pb = f2bf(p);
        prow[j] = pb;
        probs_t[((long long)(b * Sc + q) * Sc + j) * Hc + h] = pb;
    }
}

// ---------------------------------------------------------------------------
// Adjacency per-pair MLP: a[B,S,S,H] -> sigmoid(relu(a@w1+b1)@w2+b2), bf16 out
// ---------------------------------------------------------------------------
__global__ __launch_bounds__(256)
void k_adjacency(const unsigned short* __restrict__ probs_t,
                 const float* __restrict__ w1, const float* __restrict__ b1,
                 const float* __restrict__ w2, const float* __restrict__ b2,
                 unsigned short* __restrict__ adj)
{
    __shared__ float w1s[Hc * AHc];
    __shared__ float b1s[AHc];
    __shared__ float w2s[AHc];
    const int tid = threadIdx.x;
    for (int i = tid; i < Hc * AHc; i += 256) w1s[i] = w1[i];
    if (tid < AHc) { b1s[tid] = b1[tid]; w2s[tid] = w2[tid]; }
    __syncthreads();

    const long long p = (long long)blockIdx.x * 256 + tid;
    const long long total = (long long)Bc * Sc * Sc;
    if (p >= total) return;

    float a[Hc];
    const unsigned short* ar = probs_t + p * Hc;
    #pragma unroll
    for (int hh = 0; hh < Hc; ++hh) a[hh] = bf2f(ar[hh]);

    float lg = b2[0];
    #pragma unroll 4
    for (int j = 0; j < AHc; ++j) {
        float acc = b1s[j];
        #pragma unroll
        for (int hh = 0; hh < Hc; ++hh) acc = fmaf(a[hh], w1s[hh * AHc + j], acc);
        lg = fmaf(fmaxf(acc, 0.0f), w2s[j], lg);
    }
    adj[p] = f2bf(sigmoidf(lg));
}

// ---------------------------------------------------------------------------
// Merge heads: [B,H,S,HD] f32 -> [B,S,H*HD] bf16
// ---------------------------------------------------------------------------
__global__ void k_merge_heads(const float* __restrict__ src,
                              unsigned short* __restrict__ dst)
{
    const long long idx = (long long)blockIdx.x * blockDim.x + threadIdx.x;
    const long long total = (long long)Bc * Hc * Sc * HDc;
    if (idx >= total) return;
    const int d = (int)(idx % HDc);
    const int s = (int)((idx / HDc) % Sc);
    const int h = (int)((idx / ((long long)HDc * Sc)) % Hc);
    const int b = (int)(idx / ((long long)HDc * Sc * Hc));
    dst[((long long)(b * Sc + s)) * (Hc * HDc) + h * HDc + d] = f2bf(src[idx]);
}

// ---------------------------------------------------------------------------
// Elementwise epilogues
// ---------------------------------------------------------------------------
__global__ void k_add(float* __restrict__ out, const float* __restrict__ a,
                      const float* __restrict__ b, long long n)
{
    long long i = (long long)blockIdx.x * blockDim.x + threadIdx.x;
    if (i < n) out[i] = a[i] + b[i];
}
// h1 = a + b (f32 [B,S,D]); h1bt = bf16 transposed per batch -> [B,D,S]
__global__ void k_add_cast_t(float* __restrict__ out, unsigned short* __restrict__ outbt,
                             const float* __restrict__ a, const float* __restrict__ b)
{
    const long long i = (long long)blockIdx.x * blockDim.x + threadIdx.x;
    const long long total = (long long)Bc * Sc * Dc;
    if (i >= total) return;
    const int d  = (int)(i % Dc);
    const int s  = (int)((i / Dc) % Sc);
    const int bb = (int)(i / ((long long)Dc * Sc));
    const float v = a[i] + b[i];
    out[i] = v;
    outbt[((long long)bb * Dc + d) * Sc + s] = f2bf(v);
}
__global__ void k_silu_add(float* __restrict__ out, const float* __restrict__ base,
                           const float* __restrict__ g, long long n)
{
    long long i = (long long)blockIdx.x * blockDim.x + threadIdx.x;
    if (i < n) { float x = g[i]; out[i] = base[i] + x * sigmoidf(x); }
}
__global__ void k_silu_cast(unsigned short* __restrict__ dst, const float* __restrict__ src,
                            long long n)
{
    long long i = (long long)blockIdx.x * blockDim.x + threadIdx.x;
    if (i < n) { float x = src[i]; dst[i] = f2bf(x * sigmoidf(x)); }
}

// ---------------------------------------------------------------------------
// Host orchestration
// ---------------------------------------------------------------------------
static inline void gemm(const void* A, const void* B, float* C,
                        int M, int N, int K, int lda, int ldb, int ldc,
                        long long sA, long long sB, long long sC, int batches,
                        int cmode, float alpha, hipStream_t st)
{
    dim3 grid((N + 255) / 256, (M + 127) / 128, batches);
    k_gemm_bf16<<<grid, 256, 0, st>>>((const unsigned short*)A,
                                      (const unsigned short*)B, C,
                                      M, N, K, lda, ldb, ldc, sA, sB, sC, cmode, alpha);
}

extern "C" void kernel_launch(void* const* d_in, const int* in_sizes, int n_in,
                              void* d_out, int out_size, void* d_ws, size_t ws_size,
                              hipStream_t stream)
{
    (void)in_sizes; (void)n_in; (void)out_size; (void)ws_size;
    const float* hidden   = (const float*)d_in[0];
    const float* w_q      = (const float*)d_in[1];
    const float* w_k      = (const float*)d_in[2];
    const float* w_v      = (const float*)d_in[3];
    const float* w_o      = (const float*)d_in[4];
    const float* norm1_w  = (const float*)d_in[5];
    const float* norm2_w  = (const float*)d_in[6];
    const float* w_adj1   = (const float*)d_in[7];
    const float* b_adj1   = (const float*)d_in[8];
    const float* w_adj2   = (const float*)d_in[9];
    const float* b_adj2   = (const float*)d_in[10];
    const float* w_gnn    = (const float*)d_in[11];
    const float* w_mlp_in = (const float*)d_in[12];
    const float* w_mlp_out= (const float*)d_in[13];
    float* out = (float*)d_out;

    const long long NS = (long long)Bc * Sc;          // 2048 rows
    const long long DD = Dc;

    // ---- workspace bump allocator ----
    char* base = (char*)d_ws;
    size_t off = 0;
    auto alloc = [&](size_t bytes) -> char* {
        char* p = base + off;
        off = (off + bytes + 255) & ~(size_t)255;
        return p;
    };
    // all weight copies are TRANSPOSED to [N,K] bf16
    unsigned short* wqbT  = (unsigned short*)alloc((size_t)Dc * Dc * 2);
    unsigned short* wkbT  = (unsigned short*)alloc((size_t)Dc * Dc * 2);
    unsigned short* wvbT  = (unsigned short*)alloc((size_t)Dc * Dc * 2);
    unsigned short* wobT  = (unsigned short*)alloc((size_t)Dc * Dc * 2);
    unsigned short* wgbT  = (unsigned short*)alloc((size_t)Dc * Dc * 2);
    unsigned short* wmibT = (unsigned short*)alloc((size_t)Dc * Ic * 2);   // [I,D]
    unsigned short* wmobT = (unsigned short*)alloc((size_t)Ic * Dc * 2);   // [D,I]
    unsigned short* hb    = (unsigned short*)alloc((size_t)NS * DD * 2);
    float* qf             = (float*)alloc((size_t)NS * DD * 4);
    float* kf             = (float*)alloc((size_t)NS * DD * 4);
    float* vf             = (float*)alloc((size_t)NS * DD * 4);
    unsigned short* qb    = (unsigned short*)alloc((size_t)NS * DD * 2);   // [B,H,S,HD]
    unsigned short* kb    = (unsigned short*)alloc((size_t)NS * DD * 2);   // [B,H,S,HD]
    unsigned short* vbt   = (unsigned short*)alloc((size_t)NS * DD * 2);   // [B,H,HD,S]
    float* scores         = (float*)alloc((size_t)Bc * Hc * Sc * Sc * 4);
    unsigned short* probs   = (unsigned short*)alloc((size_t)Bc * Hc * Sc * Sc * 2);
    unsigned short* probs_t = (unsigned short*)alloc((size_t)Bc * Sc * Sc * Hc * 2);
    float* attnout        = (float*)alloc((size_t)NS * DD * 4);            // [B,H,S,HD]
    unsigned short* amrg  = (unsigned short*)alloc((size_t)NS * DD * 2);
    float* proj           = (float*)alloc((size_t)NS * DD * 4);
    float* h1             = (float*)alloc((size_t)NS * DD * 4);
    unsigned short* h1bt  = (unsigned short*)alloc((size_t)NS * DD * 2);   // [B,D,S]
    unsigned short* adjb  = (unsigned short*)alloc((size_t)Bc * Sc * Sc * 2);
    float* msg            = (float*)alloc((size_t)NS * DD * 4);
    unsigned short* msgb  = (unsigned short*)alloc((size_t)NS * DD * 2);
    float* gnn            = (float*)alloc((size_t)NS * DD * 4);
    float* h2             = (float*)alloc((size_t)NS * DD * 4);
    unsigned short* h3b   = (unsigned short*)alloc((size_t)NS * DD * 2);
    float* tmid           = (float*)alloc((size_t)NS * Ic * 4);
    unsigned short* tb    = (unsigned short*)alloc((size_t)NS * Ic * 2);
    float* umid           = (float*)alloc((size_t)NS * DD * 4);

    auto grid1d = [](long long n) { return dim3((unsigned)((n + 255) / 256)); };

    // ---- cast+transpose weights to bf16 [N,K] ----
    k_cast_t_bf16<<<grid1d((long long)Dc * Dc), 256, 0, stream>>>(w_q, wqbT, Dc, Dc);
    k_cast_t_bf16<<<grid1d((long long)Dc * Dc), 256, 0, stream>>>(w_k, wkbT, Dc, Dc);
    k_cast_t_bf16<<<grid1d((long long)Dc * Dc), 256, 0, stream>>>(w_v, wvbT, Dc, Dc);
    k_cast_t_bf16<<<grid1d((long long)Dc * Dc), 256, 0, stream>>>(w_o, wobT, Dc, Dc);
    k_cast_t_bf16<<<grid1d((long long)Dc * Dc), 256, 0, stream>>>(w_gnn, wgbT, Dc, Dc);
    k_cast_t_bf16<<<grid1d((long long)Dc * Ic), 256, 0, stream>>>(w_mlp_in, wmibT, Dc, Ic);
    k_cast_t_bf16<<<grid1d((long long)Ic * Dc), 256, 0, stream>>>(w_mlp_out, wmobT, Ic, Dc);

    // ---- norm1 + QKV projections ----
    k_rmsnorm_bf16<<<(unsigned)NS, 256, 0, stream>>>(hidden, norm1_w, hb, Dc);
    gemm(hb, wqbT, qf, (int)NS, Dc, Dc, Dc, Dc, Dc, 0, 0, 0, 1, 0, 1.0f, stream);
    gemm(hb, wkbT, kf, (int)NS, Dc, Dc, Dc, Dc, Dc, 0, 0, 0, 1, 0, 1.0f, stream);
    gemm(hb, wvbT, vf, (int)NS, Dc, Dc, Dc, Dc, Dc, 0, 0, 0, 1, 0, 1.0f, stream);

    // ---- RoPE + head-major bf16 (v transposed) ----
    k_rope<<<grid1d((long long)Bc * Hc * Sc), 256, 0, stream>>>(qf, kf, vf, qb, kb, vbt);

    // ---- scores = (q @ k^T) / 8, batched over B*H, causal tile skip ----
    gemm(qb, kb, scores, Sc, Sc, HDc, HDc, HDc, Sc,
         (long long)Sc * HDc, (long long)Sc * HDc, (long long)Sc * Sc,
         Bc * Hc, /*cmode=*/1, 0.125f, stream);

    // ---- causal softmax -> probs (both layouts) ----
    k_softmax_causal<<<(unsigned)(Bc * Hc * Sc), 256, 0, stream>>>(scores, probs, probs_t);

    // ---- attn_out = probs @ v, batched over B*H, K limited by causality ----
    gemm(probs, vbt, attnout, Sc, HDc, Sc, Sc, Sc, HDc,
         (long long)Sc * Sc, (long long)HDc * Sc, (long long)Sc * HDc,
         Bc * Hc, /*cmode=*/2, 1.0f, stream);

    // ---- merge heads, output proj, residual ----
    k_merge_heads<<<grid1d((long long)NS * DD), 256, 0, stream>>>(attnout, amrg);
    gemm(amrg, wobT, proj, (int)NS, Dc, Dc, Dc, Dc, Dc, 0, 0, 0, 1, 0, 1.0f, stream);
    k_add_cast_t<<<grid1d(NS * DD), 256, 0, stream>>>(h1, h1bt, hidden, proj);

    // ---- adjacency MLP over heads ----
    k_adjacency<<<grid1d((long long)Bc * Sc * Sc), 256, 0, stream>>>(
        probs_t, w_adj1, b_adj1, w_adj2, b_adj2, adjb);

    // ---- GNN: msg = adj @ h1 (batched over B), then silu(msg @ w_gnn) ----
    gemm(adjb, h1bt, msg, Sc, Dc, Sc, Sc, Sc, Dc,
         (long long)Sc * Sc, (long long)Dc * Sc, (long long)Sc * Dc,
         Bc, 0, 1.0f, stream);
    k_cast_bf16<<<grid1d(NS * DD), 256, 0, stream>>>(msg, msgb, NS * DD);
    gemm(msgb, wgbT, gnn, (int)NS, Dc, Dc, Dc, Dc, Dc, 0, 0, 0, 1, 0, 1.0f, stream);
    k_silu_add<<<grid1d(NS * DD), 256, 0, stream>>>(h2, h1, gnn, NS * DD);

    // ---- norm2 + MLP + final residual ----
    k_rmsnorm_bf16<<<(unsigned)NS, 256, 0, stream>>>(h2, norm2_w, h3b, Dc);
    gemm(h3b, wmibT, tmid, (int)NS, Ic, Dc, Dc, Dc, Ic, 0, 0, 0, 1, 0, 1.0f, stream);
    k_silu_cast<<<grid1d(NS * (long long)Ic), 256, 0, stream>>>(tb, tmid, NS * (long long)Ic);
    gemm(tb, wmobT, umid, (int)NS, Dc, Ic, Ic, Ic, Dc, 0, 0, 0, 1, 0, 1.0f, stream);
    k_add<<<grid1d(NS * DD), 256, 0, stream>>>(out, h2, umid, NS * DD);
}
